// MultiHeadAttention_19421842112845
// MI455X (gfx1250) — matrix operand order repaired and checked by reading, not compiled, not run on hardware
//
#include <hip/hip_runtime.h>
#include <hip/hip_bf16.h>
#include <math.h>

#define BB 4
#define SS 2048
#define DD 1024
#define HH 16
#define DH 64

typedef __attribute__((ext_vector_type(16))) _Float16 v16h;
typedef __attribute__((ext_vector_type(8)))  _Float16 v8h;
typedef __attribute__((ext_vector_type(8)))  float    v8f;
typedef __attribute__((ext_vector_type(4)))  float    v4f;

union H16 { v16h v; v8h h[2]; };

__device__ __forceinline__ v8f wmma_f16(const H16& a, const H16& b, v8f c) {
  // D = A(16x32 f16) * B(32x16 f16) + C(f32)
  return __builtin_amdgcn_wmma_f32_16x16x32_f16(false, a.v, false, b.v,
                                                (short)0, c, false, false);
}

// ---------------------------------------------------------------------------
// Kernel 1: per-head QKV projection.  grid = (S/16, B*H, 3), block = 128.
// Each wave computes one 16x16 tile of [16 x DH]; K-loop over D in chunks of 32.
// Q,K stored [bh][s][DH] f16; V stored transposed [bh][DH][s] f16 so the
// attention kernel's PV B-fragments are contiguous.
// ---------------------------------------------------------------------------
__global__ __launch_bounds__(128) void mha_qkv_proj(
    const float* __restrict__ Xq, const float* __restrict__ Xk,
    const float* __restrict__ Xv,
    const float* __restrict__ Wq, const float* __restrict__ Wk,
    const float* __restrict__ Wv,
    _Float16* __restrict__ Qh, _Float16* __restrict__ Kh,
    _Float16* __restrict__ Vt)
{
  const int mat  = blockIdx.z;
  const float* X = (mat == 0) ? Xq : (mat == 1) ? Xk : Xv;
  const float* W = (mat == 0) ? Wq : (mat == 1) ? Wk : Wv;
  const int bh   = blockIdx.y;
  const int b    = bh >> 4;
  const int h    = bh & (HH - 1);
  const int s0   = blockIdx.x * 16;
  const int wave = threadIdx.x >> 5;
  const int lane = threadIdx.x & 31;
  const int lh   = lane >> 4;      // half-wave select
  const int lr   = lane & 15;      // row/col within 16
  const int n0   = wave * 16;      // this wave's N tile within DH

  const float* Xrow = X + ((size_t)b * SS + (s0 + lr)) * DD;
  const float* Wm   = W + (size_t)h * DD * DH;

  v8f acc = (v8f){0.f,0.f,0.f,0.f,0.f,0.f,0.f,0.f};
  for (int k0 = 0; k0 < DD; k0 += 32) {
    // A fragment: row m = lr, halves 0..7 -> k = lh*8+0..7, 8..15 -> k = 16+lh*8+0..7
    H16 a;
    {
      const v4f x0 = *(const v4f*)(Xrow + k0 + lh*8);
      const v4f x1 = *(const v4f*)(Xrow + k0 + lh*8 + 4);
      const v4f x2 = *(const v4f*)(Xrow + k0 + lh*8 + 16);
      const v4f x3 = *(const v4f*)(Xrow + k0 + lh*8 + 20);
      #pragma unroll
      for (int i = 0; i < 4; ++i) {
        a.h[0][i]   = (_Float16)x0[i];
        a.h[0][i+4] = (_Float16)x1[i];
        a.h[1][i]   = (_Float16)x2[i];
        a.h[1][i+4] = (_Float16)x3[i];
      }
    }
    // B fragment: col n = n0+lr, h[0][t] -> k = k0 + 8*lh + t, h[1][t] -> k = k0 + 16 + 8*lh + t (stride DH in W)
    H16 bf;
    {
      const float* wp = Wm + (size_t)(k0 + lh*8) * DH + (n0 + lr);
      #pragma unroll
      for (int t = 0; t < 8; ++t) {
        bf.h[0][t] = (_Float16)wp[(size_t)t * DH];
        bf.h[1][t] = (_Float16)wp[(size_t)(t + 16) * DH];
      }
    }
    acc = wmma_f16(a, bf, acc);
  }

  if (mat < 2) {
    // C layout: lane holds rows m = lh*8+v, col = n0+lr
    _Float16* O = ((mat == 0) ? Qh : Kh) +
                  ((size_t)bh * SS + s0) * DH + n0 + lr;
    #pragma unroll
    for (int vv = 0; vv < 8; ++vv)
      O[(size_t)(lh*8 + vv) * DH] = (_Float16)acc[vv];
  } else {
    // transposed store: Vt[bh][d = n0+lr][s0 + lh*8 + 0..7] -> contiguous 16B
    v8h pk;
    #pragma unroll
    for (int vv = 0; vv < 8; ++vv) pk[vv] = (_Float16)acc[vv];
    *(v8h*)(Vt + ((size_t)bh * DH + n0 + lr) * SS + s0 + lh*8) = pk;
  }
}

// ---------------------------------------------------------------------------
// Kernel 2: flash attention.  grid = (S/64, B*H), block = 128 (4 waves);
// each wave owns one 16-query tile, iterates 32-key chunks with online softmax.
// ---------------------------------------------------------------------------
__global__ __launch_bounds__(128) void mha_flash_attn(
    const _Float16* __restrict__ Qh, const _Float16* __restrict__ Kh,
    const _Float16* __restrict__ Vt, _Float16* __restrict__ Oh,
    const int* __restrict__ causal_flag)
{
  __shared__ alignas(16) _Float16 pbuf[4][16 * 32];   // wave-private P tiles
  const int bh   = blockIdx.y;
  const int wave = threadIdx.x >> 5;
  const int lane = threadIdx.x & 31;
  const int lh   = lane >> 4;
  const int lr   = lane & 15;
  const int q0   = (blockIdx.x * 4 + wave) * 16;
  const int causal = causal_flag[0];

  const _Float16* Qb = Qh + (size_t)bh * SS * DH;
  const _Float16* Kb = Kh + (size_t)bh * SS * DH;
  const _Float16* Vb = Vt + (size_t)bh * DH * SS;

  // Q fragments for d-chunks [0,32) and [32,64); held in registers all loop.
  H16 A0, A1;
  {
    const _Float16* qrow = Qb + (size_t)(q0 + lr) * DH + lh*16;
    A0.h[0] = *(const v8h*)(qrow);
    A0.h[1] = *(const v8h*)(qrow + 8);
    A1.h[0] = *(const v8h*)(qrow + 32);
    A1.h[1] = *(const v8h*)(qrow + 40);
  }

  v8f accN[4];
  #pragma unroll
  for (int nt = 0; nt < 4; ++nt)
    accN[nt] = (v8f){0.f,0.f,0.f,0.f,0.f,0.f,0.f,0.f};
  float mrow[8], lsum[8];
  #pragma unroll
  for (int vv = 0; vv < 8; ++vv) { mrow[vv] = -INFINITY; lsum[vv] = 0.f; }

  const int jend = causal ? q0 : (SS - 32);
  for (int j0 = 0; j0 <= jend; j0 += 32) {
    // ---- scores: two 16x16 N-tiles, each accumulated over two d-chunks ----
    v8f sc[2];
    #pragma unroll
    for (int t = 0; t < 2; ++t) {
      const _Float16* krow = Kb + (size_t)(j0 + 16*t + lr) * DH + lh*16;
      H16 B0, B1;                          // B[k=d][n=key]: contiguous in d
      B0.h[0] = *(const v8h*)(krow);
      B0.h[1] = *(const v8h*)(krow + 8);
      B1.h[0] = *(const v8h*)(krow + 32);
      B1.h[1] = *(const v8h*)(krow + 40);
      v8f c = (v8f){0.f,0.f,0.f,0.f,0.f,0.f,0.f,0.f};
      c = wmma_f16(A0, B0, c);
      c = wmma_f16(A1, B1, c);
      sc[t] = c;
    }

    // ---- scale + causal mask + per-row max ----
    float mx[8], rs[8];
    #pragma unroll
    for (int vv = 0; vv < 8; ++vv) {
      float sv0 = sc[0][vv] * 0.125f;      // 1/sqrt(DH)
      float sv1 = sc[1][vv] * 0.125f;
      if (causal) {
        const int row = q0 + lh*8 + vv;
        if (j0 + lr      > row) sv0 = -INFINITY;
        if (j0 + 16 + lr > row) sv1 = -INFINITY;
      }
      sc[0][vv] = sv0; sc[1][vv] = sv1;
      mx[vv] = fmaxf(sv0, sv1);
    }
    #pragma unroll
    for (int vv = 0; vv < 8; ++vv) {
      #pragma unroll
      for (int off = 1; off < 16; off <<= 1)
        mx[vv] = fmaxf(mx[vv], __shfl_xor(mx[vv], off, 32));
    }

    // ---- online softmax update, write P (f16) into wave-private LDS ----
    #pragma unroll
    for (int vv = 0; vv < 8; ++vv) {
      const float m_new = fmaxf(mrow[vv], mx[vv]);
      const float sf    = __expf(mrow[vv] - m_new);
      mrow[vv]  = m_new;
      lsum[vv] *= sf;
      #pragma unroll
      for (int nt = 0; nt < 4; ++nt) accN[nt][vv] *= sf;
      const float p0 = __expf(sc[0][vv] - m_new);
      const float p1 = __expf(sc[1][vv] - m_new);
      rs[vv] = p0 + p1;
      const int row8 = lh*8 + vv;
      pbuf[wave][row8*32 + lr]      = (_Float16)p0;
      pbuf[wave][row8*32 + 16 + lr] = (_Float16)p1;
    }
    #pragma unroll
    for (int vv = 0; vv < 8; ++vv) {
      #pragma unroll
      for (int off = 1; off < 16; off <<= 1)
        rs[vv] += __shfl_xor(rs[vv], off, 32);
      lsum[vv] += rs[vv];
    }

    // LDS is in-order per wave; wait for the cross-lane P stores to land.
    asm volatile("s_wait_dscnt 0x0" ::: "memory");

    // ---- re-shape P: C-layout -> A-layout via LDS ----
    H16 Ap;
    {
      const _Float16* pb = &pbuf[wave][lr*32 + lh*8];
      Ap.h[0] = *(const v8h*)(pb);
      Ap.h[1] = *(const v8h*)(pb + 16);
    }

    // ---- acc += P(16x32) * V(32x64), V read from transposed layout ----
    #pragma unroll
    for (int nt = 0; nt < 4; ++nt) {
      const _Float16* vp = Vb + (size_t)(nt*16 + lr) * SS + j0 + lh*8;
      H16 Bv;                              // B[k=key][n=d]: contiguous in key
      Bv.h[0] = *(const v8h*)(vp);
      Bv.h[1] = *(const v8h*)(vp + 16);
      accN[nt] = wmma_f16(Ap, Bv, accN[nt]);
    }
  }

  // ---- normalize and store f16 head output [bh][s][DH] ----
  _Float16* Orow = Oh + ((size_t)bh * SS + q0) * DH;
  #pragma unroll
  for (int vv = 0; vv < 8; ++vv) {
    const float inv = 1.0f / lsum[vv];
    #pragma unroll
    for (int nt = 0; nt < 4; ++nt)
      Orow[(size_t)(lh*8 + vv) * DH + nt*16 + lr] =
          (_Float16)(accN[nt][vv] * inv);
  }
}

// ---------------------------------------------------------------------------
// Kernel 3: output projection out = Xo @ Wo^T + bo.
// grid = (B*S/16, D/64), block = 128; one 16x16 tile per wave.
// ---------------------------------------------------------------------------
__global__ __launch_bounds__(128) void mha_out_proj(
    const _Float16* __restrict__ Oh, const float* __restrict__ Wo,
    const float* __restrict__ bo, float* __restrict__ out)
{
  const int st   = blockIdx.x;             // 16-row tile over B*S
  const int wave = threadIdx.x >> 5;
  const int lane = threadIdx.x & 31;
  const int lh = lane >> 4, lr = lane & 15;
  const int n0 = blockIdx.y * 64 + wave * 16;

  const int srow = st * 16 + lr;           // this lane's A row (global)
  const int b = srow / SS;
  const int s = srow - b * SS;

  v8f acc = (v8f){0.f,0.f,0.f,0.f,0.f,0.f,0.f,0.f};
  for (int k0 = 0; k0 < DD; k0 += 32) {
    // A from head-interleaved f16 workspace: i = h*64 + d, 32-chunk stays in head
    const int head = k0 >> 6;
    const int dk   = (k0 & 63) + lh*16;
    const _Float16* xr = Oh + (((size_t)b * HH + head) * SS + s) * DH + dk;
    H16 a;
    a.h[0] = *(const v8h*)(xr);
    a.h[1] = *(const v8h*)(xr + 8);
    // B[k=i][n=j] = Wo[j][i]: 16 contiguous fp32 per lane, cvt to f16
    const float* wr = Wo + (size_t)(n0 + lr) * DD + k0 + lh*16;
    const v4f w0 = *(const v4f*)(wr);
    const v4f w1 = *(const v4f*)(wr + 4);
    const v4f w2 = *(const v4f*)(wr + 8);
    const v4f w3 = *(const v4f*)(wr + 12);
    H16 bf;
    #pragma unroll
    for (int i = 0; i < 4; ++i) {
      bf.h[0][i]   = (_Float16)w0[i];
      bf.h[0][i+4] = (_Float16)w1[i];
      bf.h[1][i]   = (_Float16)w2[i];
      bf.h[1][i+4] = (_Float16)w3[i];
    }
    acc = wmma_f16(a, bf, acc);
  }

  const int j = n0 + lr;
  const float bj = bo[j];
  float* obase = out + (size_t)st * 16 * DD + j;
  #pragma unroll
  for (int vv = 0; vv < 8; ++vv)
    obase[(size_t)(lh*8 + vv) * DD] = acc[vv] + bj;
}

// ---------------------------------------------------------------------------
extern "C" void kernel_launch(void* const* d_in, const int* in_sizes, int n_in,
                              void* d_out, int out_size, void* d_ws, size_t ws_size,
                              hipStream_t stream) {
  (void)in_sizes; (void)n_in; (void)out_size; (void)ws_size;
  const float* q   = (const float*)d_in[0];
  const float* k   = (const float*)d_in[1];
  const float* v   = (const float*)d_in[2];
  const float* Wq  = (const float*)d_in[3];
  const float* Wk  = (const float*)d_in[4];
  const float* Wv  = (const float*)d_in[5];
  const float* Wo  = (const float*)d_in[6];
  const float* bo  = (const float*)d_in[7];
  const int* causal = (const int*)d_in[8];
  float* out = (float*)d_out;

  const size_t per = (size_t)BB * HH * SS * DH;   // 8,388,608 f16 elements
  _Float16* Qh = (_Float16*)d_ws;                  // [bh][s][dh]
  _Float16* Kh = Qh + per;                         // [bh][s][dh]
  _Float16* Vt = Kh + per;                         // [bh][dh][s] (transposed)
  _Float16* Oh = Vt + per;                         // [bh][s][dh]

  mha_qkv_proj<<<dim3(SS/16, BB*HH, 3), 128, 0, stream>>>(
      q, k, v, Wq, Wk, Wv, Qh, Kh, Vt);
  mha_flash_attn<<<dim3(SS/64, BB*HH), 128, 0, stream>>>(
      Qh, Kh, Vt, Oh, causal);
  mha_out_proj<<<dim3(BB*SS/16, DD/64), 128, 0, stream>>>(
      Oh, Wo, bo, out);
}
